// SeparateModalityEncoderMOE_44676249813387
// MI455X (gfx1250) — compile-verified, matrix operand order checked by reading, not run on hardware
//
#include <hip/hip_runtime.h>
#include <hip/hip_bf16.h>

#define EE   32
#define BB   16
#define TT   512
#define CTXC 32
#define GG   8
#define HH   128

typedef __attribute__((ext_vector_type(16))) _Float16 v16h;
typedef __attribute__((ext_vector_type(8)))  _Float16 v8h;
typedef __attribute__((ext_vector_type(8)))  float    v8f;

union FragH { v16h v; v8h h8[2]; _Float16 h[16]; };
union FragF { v8f  v; float f[8]; };

// K-offset within a 32-wide chunk for 16-bit WMMA A/B fragments (ISA 7.12.2):
// lanes 0-15 hold K {0..7,16..23}, lanes 16-31 hold K {8..15,24..31}.
__device__ __forceinline__ int wmma_koff(int i, int hi) {
    int v = i >> 1, p = i & 1;
    int base = (v < 4) ? (v << 1) : (8 + (v << 1));
    return base + (hi << 3) + p;
}

__device__ __forceinline__ float sigf(float x) { return 1.0f / (1.0f + expf(-x)); }

// ---------------------------------------------------------------- casts
__global__ void k_cast_f16(const float* __restrict__ in, _Float16* __restrict__ out, long long n) {
    long long i = (long long)blockIdx.x * blockDim.x + threadIdx.x;
    long long stride = (long long)gridDim.x * blockDim.x;
    for (; i < n; i += stride) out[i] = (_Float16)in[i];
}

// cast + pad rows: out[row][0..KlenP) = in[row][0..Klen) then zeros
__global__ void k_cast_pad(const float* __restrict__ in, _Float16* __restrict__ out,
                           int rows, int Klen, int KlenP) {
    long long n = (long long)rows * KlenP;
    long long i = (long long)blockIdx.x * blockDim.x + threadIdx.x;
    long long stride = (long long)gridDim.x * blockDim.x;
    for (; i < n; i += stride) {
        int row = i / KlenP;
        int k = i - (long long)row * KlenP;
        out[i] = (k < Klen) ? (_Float16)in[(long long)row * Klen + k] : (_Float16)0.f;
    }
}

// ---------------------------------------------------------------- context conv
__global__ void k_ctx_conv(const float* __restrict__ xe, const float* __restrict__ xi,
                           const float* __restrict__ cw, const float* __restrict__ cb,
                           float* __restrict__ y) {
    long long gid = (long long)blockIdx.x * blockDim.x + threadIdx.x; // B*32*T
    if (gid >= (long long)BB * CTXC * TT) return;
    int t = gid % TT;
    int c = (gid / TT) % CTXC;
    int b = gid / ((long long)TT * CTXC);
    float acc = cb[c];
    for (int ci = 0; ci < 88; ci++) {
        const float* xs = (ci < 16) ? (xe + ((long long)b * 16 + ci) * TT)
                                    : (xi + ((long long)b * 72 + (ci - 16)) * TT);
        const float* wp = cw + ((long long)c * 88 + ci) * 5;
#pragma unroll
        for (int kk = 0; kk < 5; kk++) {
            int tt = t + kk - 2;
            if (tt >= 0 && tt < TT) acc += wp[kk] * xs[tt];
        }
    }
    y[gid] = acc;
}

// ---------------------------------------------------------------- context GN + relu + mean_t -> rbar[B][32]
__global__ void k_ctx_gn(const float* __restrict__ y, const float* __restrict__ gs,
                         const float* __restrict__ gb, float* __restrict__ rbar) {
    int g = blockIdx.x & 7;
    int b = blockIdx.x >> 3;
    const float* yb = y + ((long long)b * CTXC + g * 4) * TT;
    __shared__ float red[256];
    __shared__ float red2[256];
    float s = 0.f, s2 = 0.f;
    for (int i = threadIdx.x; i < 4 * TT; i += 256) { float v = yb[i]; s += v; s2 += v * v; }
    red[threadIdx.x] = s; red2[threadIdx.x] = s2; __syncthreads();
    for (int st = 128; st > 0; st >>= 1) {
        if (threadIdx.x < st) { red[threadIdx.x] += red[threadIdx.x + st]; red2[threadIdx.x] += red2[threadIdx.x + st]; }
        __syncthreads();
    }
    float mu = red[0] / 2048.f;
    float var = red2[0] / 2048.f - mu * mu;
    float rinv = rsqrtf(var + 1e-5f);
    __syncthreads();
    for (int c = 0; c < 4; c++) {
        int ch = g * 4 + c;
        float sc = gs[ch], bi = gb[ch];
        float ps = 0.f;
        for (int t = threadIdx.x; t < TT; t += 256) {
            float v = (yb[c * TT + t] - mu) * rinv * sc + bi;
            ps += (v > 0.f) ? v : 0.f;
        }
        red[threadIdx.x] = ps; __syncthreads();
        for (int st = 128; st > 0; st >>= 1) {
            if (threadIdx.x < st) red[threadIdx.x] += red[threadIdx.x + st];
            __syncthreads();
        }
        if (threadIdx.x == 0) rbar[b * CTXC + ch] = red[0] / (float)TT;
        __syncthreads();
    }
}

// ---------------------------------------------------------------- gate
__global__ void k_gate(const float* __restrict__ rbar, const float* __restrict__ lw,
                       const float* __restrict__ lb, const float* __restrict__ gw,
                       const float* __restrict__ gbv, float* __restrict__ w_hard) {
    int b = threadIdx.x;
    if (b >= BB) return;
    float r2[CTXC];
    for (int c = 0; c < CTXC; c++) {
        float s = lb[c];
        for (int k = 0; k < CTXC; k++) s += rbar[b * CTXC + k] * lw[c * CTXC + k];
        r2[c] = s;
    }
    float p[EE]; float mx = -1e30f;
    for (int e = 0; e < EE; e++) {
        float s = gbv[e];
        for (int c = 0; c < CTXC; c++) s += r2[c] * gw[e * CTXC + c];
        p[e] = s / 0.65f;
        if (p[e] > mx) mx = p[e];
    }
    float se = 0.f;
    for (int e = 0; e < EE; e++) { p[e] = expf(p[e] - mx); se += p[e]; }
    for (int e = 0; e < EE; e++) p[e] /= se;
    bool sel[EE];
    for (int e = 0; e < EE; e++) sel[e] = false;
    float ssum = 0.f;
    for (int it = 0; it < 9; it++) {
        int am = -1; float mv = -1.f;
        for (int e = 0; e < EE; e++)
            if (!sel[e] && p[e] > mv) { mv = p[e]; am = e; }
        sel[am] = true; ssum += mv;
    }
    for (int e = 0; e < EE; e++) w_hard[b * EE + e] = sel[e] ? (p[e] / ssum) : 0.f;
}

// ---------------------------------------------------------------- expert conv as implicit GEMM via WMMA
// weights pre-cast to f16, rows padded to KlenP -> A fragment = 2x b128 loads/lane
template <int Cin, int Cout>
__global__ __launch_bounds__(256) void k_conv_wmma(
    const _Float16* __restrict__ xin, long long xin_estride,
    const _Float16* __restrict__ wf, const float* __restrict__ bias,
    float* __restrict__ y) {
    constexpr int Klen  = Cin * 5;
    constexpr int nK    = (Klen + 31) / 32;
    constexpr int KlenP = nK * 32;
    constexpr int ntT   = TT / 16;
    constexpr int ntM   = Cout / 16;
    int wg = blockIdx.x * 8 + (threadIdx.x >> 5);
    int nt = wg % ntT;  wg /= ntT;
    int mt = wg % ntM;  wg /= ntM;
    int b  = wg % BB;
    int e  = wg / BB;
    int lane = threadIdx.x & 31;
    int l = lane & 15, hi = lane >> 4;
    const _Float16* xb = xin + (long long)e * xin_estride + (long long)b * Cin * TT;
    const _Float16* arow = wf + (long long)(e * Cout + mt * 16 + l) * KlenP;
    int t0 = nt * 16;
    FragF acc;
#pragma unroll
    for (int r = 0; r < 8; r++) acc.f[r] = 0.f;
#pragma unroll
    for (int kc = 0; kc < nK; kc++) {
        int kb = kc * 32;
        FragH A, Bm;
        A.h8[0] = *(const v8h*)(arow + kb + 8 * hi);
        A.h8[1] = *(const v8h*)(arow + kb + 16 + 8 * hi);
#pragma unroll
        for (int i = 0; i < 16; i++) {
            int k = kb + wmma_koff(i, hi);
            _Float16 xv = (_Float16)0.f;
            if (k < Klen) {
                int ci = k / 5, kk = k - ci * 5;
                int t = t0 + l + kk - 2;
                if ((unsigned)t < (unsigned)TT) xv = xb[(long long)ci * TT + t];
            }
            Bm.h[i] = xv;
        }
        acc.v = __builtin_amdgcn_wmma_f32_16x16x32_f16(false, A.v, false, Bm.v,
                                                       (short)0, acc.v, false, false);
    }
    float* yb = y + (long long)(e * BB + b) * Cout * TT;
#pragma unroll
    for (int r = 0; r < 8; r++) {
        int co = mt * 16 + r + hi * 8;
        yb[(long long)co * TT + t0 + l] = acc.f[r] + bias[e * Cout + co];
    }
}

// ---------------------------------------------------------------- GroupNorm + ReLU -> f16 activations
__global__ void k_gn_relu(const float* __restrict__ y, const float* __restrict__ gs,
                          const float* __restrict__ gb, _Float16* __restrict__ act,
                          int C, int Ctot, int co_off) {
    int g = blockIdx.x % GG;
    int tmp = blockIdx.x / GG;
    int b = tmp % BB;
    int e = tmp / BB;
    int Cg = C / GG;
    const float* yb = y + ((long long)(e * BB + b) * C + g * Cg) * TT;
    int n = Cg * TT;
    __shared__ float red[256];
    __shared__ float red2[256];
    float s = 0.f, s2 = 0.f;
    for (int i = threadIdx.x; i < n; i += 256) { float v = yb[i]; s += v; s2 += v * v; }
    red[threadIdx.x] = s; red2[threadIdx.x] = s2; __syncthreads();
    for (int st = 128; st > 0; st >>= 1) {
        if (threadIdx.x < st) { red[threadIdx.x] += red[threadIdx.x + st]; red2[threadIdx.x] += red2[threadIdx.x + st]; }
        __syncthreads();
    }
    float mu = red[0] / (float)n;
    float var = red2[0] / (float)n - mu * mu;
    float rinv = rsqrtf(var + 1e-5f);
    _Float16* ab = act + ((long long)(e * BB + b) * Ctot + co_off + g * Cg) * TT;
    for (int i = threadIdx.x; i < n; i += 256) {
        int c = i / TT;
        float sc = gs[e * C + g * Cg + c];
        float bi = gb[e * C + g * Cg + c];
        float v = (yb[i] - mu) * rinv * sc + bi;
        ab[i] = (_Float16)((v > 0.f) ? v : 0.f);
    }
}

// ---------------------------------------------------------------- weighted expert combine -> h[B][T][256] f16
// thread handles 8 consecutive t for one (b,c): vectorized 16B reads of stacked
__global__ void k_combine(const _Float16* __restrict__ stacked, const float* __restrict__ w_hard,
                          _Float16* __restrict__ h) {
    long long gid = (long long)blockIdx.x * 256 + threadIdx.x; // B*256*(T/8)
    int tc = gid % (TT / 8);
    long long rest = gid / (TT / 8);
    int c = rest % 256;
    int b = rest / 256;
    int t0 = tc * 8;
    float acc[8];
#pragma unroll
    for (int j = 0; j < 8; j++) acc[j] = 0.f;
    for (int e = 0; e < EE; e++) {
        float wv = w_hard[b * EE + e];
        if (wv != 0.f) {
            v8h xv = *(const v8h*)(stacked + (((long long)(e * BB + b) * 256 + c) * TT + t0));
#pragma unroll
            for (int j = 0; j < 8; j++) acc[j] += wv * (float)xv[j];
        }
    }
#pragma unroll
    for (int j = 0; j < 8; j++)
        h[((long long)(b * TT + t0 + j)) * 256 + c] = (_Float16)acc[j];
}

// ---------------------------------------------------------------- xproj = h @ Wih^T + bih + bhh  (M=8192,N=512,K=256)
__global__ __launch_bounds__(256) void k_xproj(
    const _Float16* __restrict__ h, const _Float16* __restrict__ Wih_base,
    const float* __restrict__ bih_base, const float* __restrict__ bhh_base,
    float* __restrict__ xp_base) {
    int bid = blockIdx.x;
    int d = bid >> 11;          // 2048 blocks per direction
    bid &= 2047;
    const _Float16* Wih = Wih_base + (long long)d * 512 * 256;
    const float* bih = bih_base + d * 512;
    const float* bhh = bhh_base + d * 512;
    float* xp = xp_base + (long long)d * 8192 * 512;

    int wg = bid * 8 + (threadIdx.x >> 5);
    int ntile = wg % 32;
    int mtile = wg / 32;        // 0..511
    int lane = threadIdx.x & 31;
    int l = lane & 15, hi = lane >> 4;
    const _Float16* arow = h + (long long)(mtile * 16 + l) * 256;
    const _Float16* brow = Wih + (long long)(ntile * 16 + l) * 256;
    FragF acc;
#pragma unroll
    for (int r = 0; r < 8; r++) acc.f[r] = 0.f;
#pragma unroll
    for (int kc = 0; kc < 8; kc++) {   // K=256
        int kb = kc * 32;
        FragH A, Bm;
        A.h8[0] = *(const v8h*)(arow + kb + 8 * hi);
        A.h8[1] = *(const v8h*)(arow + kb + 16 + 8 * hi);
        Bm.h8[0] = *(const v8h*)(brow + kb + 8 * hi);
        Bm.h8[1] = *(const v8h*)(brow + kb + 16 + 8 * hi);
        acc.v = __builtin_amdgcn_wmma_f32_16x16x32_f16(false, A.v, false, Bm.v,
                                                       (short)0, acc.v, false, false);
    }
#pragma unroll
    for (int r = 0; r < 8; r++) {
        int m = mtile * 16 + r + hi * 8;
        int n = ntile * 16 + l;
        xp[(long long)m * 512 + n] = acc.f[r] + bih[n] + bhh[n];
    }
}

// ---------------------------------------------------------------- LSTM recurrence: one block per direction
// Whh fragments held in registers across all 512 steps; h lives in LDS.
__global__ __launch_bounds__(512) void k_lstm_rec(
    const float* __restrict__ xp_base, const _Float16* __restrict__ Whh_base,
    _Float16* __restrict__ hout) {
    int d = blockIdx.x;
    const float* xp = xp_base + (long long)d * 8192 * 512;
    const _Float16* Whh = Whh_base + (long long)d * 512 * 128;
    int dir_off = d * 128;
    int rev = d;

    __shared__ __align__(16) _Float16 hlds[16 * 128];
    __shared__ float glds[16 * 512];
    int tid = threadIdx.x;
    int wv = tid >> 5, lane = tid & 31, l = lane & 15, hi = lane >> 4;
    float cst[4];
    int eb[4], eh[4];
#pragma unroll
    for (int j = 0; j < 4; j++) {
        cst[j] = 0.f;
        int idx = tid * 4 + j;
        eb[j] = idx >> 7;
        eh[j] = idx & 127;
    }
    for (int i = tid; i < 16 * 128; i += 512) hlds[i] = (_Float16)0.f;

    // preload Whh B-fragments into registers: 2 tiles x 4 K-chunks
    FragH Bfr[2][4];
#pragma unroll
    for (int q = 0; q < 2; q++) {
        int ntile = wv * 2 + q;
        const _Float16* brow = Whh + (long long)(ntile * 16 + l) * 128;
#pragma unroll
        for (int kc = 0; kc < 4; kc++) {
            int kb = kc * 32;
            Bfr[q][kc].h8[0] = *(const v8h*)(brow + kb + 8 * hi);
            Bfr[q][kc].h8[1] = *(const v8h*)(brow + kb + 16 + 8 * hi);
        }
    }
    __syncthreads();

    for (int s = 0; s < TT; s++) {
        int t = rev ? (TT - 1 - s) : s;
        // A fragments from LDS h (shared across both N tiles)
        FragH A[4];
        const _Float16* hrow = hlds + l * 128;
#pragma unroll
        for (int kc = 0; kc < 4; kc++) {
            int kb = kc * 32;
            A[kc].h8[0] = *(const v8h*)(hrow + kb + 8 * hi);
            A[kc].h8[1] = *(const v8h*)(hrow + kb + 16 + 8 * hi);
        }
#pragma unroll
        for (int q = 0; q < 2; q++) {
            int ntile = wv * 2 + q;
            FragF acc;
#pragma unroll
            for (int r = 0; r < 8; r++) acc.f[r] = 0.f;
#pragma unroll
            for (int kc = 0; kc < 4; kc++)
                acc.v = __builtin_amdgcn_wmma_f32_16x16x32_f16(false, A[kc].v, false, Bfr[q][kc].v,
                                                               (short)0, acc.v, false, false);
#pragma unroll
            for (int r = 0; r < 8; r++) {
                int m = r + hi * 8;                 // batch row
                int n = ntile * 16 + l;             // gate index
                glds[m * 512 + n] = acc.f[r] + xp[((long long)m * TT + t) * 512 + n];
            }
        }
        __syncthreads();
#pragma unroll
        for (int j = 0; j < 4; j++) {
            float gi = glds[eb[j] * 512 + eh[j]];
            float gf = glds[eb[j] * 512 + 128 + eh[j]];
            float gg = glds[eb[j] * 512 + 256 + eh[j]];
            float go = glds[eb[j] * 512 + 384 + eh[j]];
            float c = sigf(gf) * cst[j] + sigf(gi) * tanhf(gg);
            cst[j] = c;
            float hv = sigf(go) * tanhf(c);
            hlds[eb[j] * 128 + eh[j]] = (_Float16)hv;
            hout[((long long)eb[j] * TT + t) * 256 + dir_off + eh[j]] = (_Float16)hv;
        }
        __syncthreads();
    }
}

// ---------------------------------------------------------------- head
__global__ void k_head(const _Float16* __restrict__ h3, const float* __restrict__ w1,
                       const float* __restrict__ b1, const float* __restrict__ w2,
                       const float* __restrict__ b2, float* __restrict__ out) {
    __shared__ float feat[16 * 256];
    __shared__ float z[16 * 128];
    for (int idx = threadIdx.x; idx < 16 * 256; idx += 256) {
        int b = idx >> 8, c = idx & 255;
        float s = 0.f;
        for (int t = 0; t < TT; t++) s += (float)h3[((long long)b * TT + t) * 256 + c];
        feat[idx] = s * (1.f / (float)TT);
    }
    __syncthreads();
    for (int idx = threadIdx.x; idx < 16 * 128; idx += 256) {
        int b = idx >> 7, j = idx & 127;
        float s = b1[j];
        for (int c = 0; c < 256; c++) s += feat[b * 256 + c] * w1[j * 256 + c];
        z[idx] = 0.5f * s * (1.f + erff(s * 0.7071067811865476f));
    }
    __syncthreads();
    if (threadIdx.x < 160) {
        int b = threadIdx.x / 10, j = threadIdx.x % 10;
        float s = b2[j];
        for (int k = 0; k < 128; k++) s += z[b * 128 + k] * w2[j * 128 + k];
        out[b * 10 + j] = s;
    }
}

// ================================================================ host
extern "C" void kernel_launch(void* const* d_in, const int* in_sizes, int n_in,
                              void* d_out, int out_size, void* d_ws, size_t ws_size,
                              hipStream_t stream) {
    const float* x_emg = (const float*)d_in[0];
    const float* x_imu = (const float*)d_in[1];
    const float* ctx_cw = (const float*)d_in[2];
    const float* ctx_cb = (const float*)d_in[3];
    const float* ctx_gs = (const float*)d_in[4];
    const float* ctx_gb = (const float*)d_in[5];
    const float* ctx_lw = (const float*)d_in[6];
    const float* ctx_lb = (const float*)d_in[7];
    const float* gate_w = (const float*)d_in[8];
    const float* gate_b = (const float*)d_in[9];
    const float* emg_w[3]  = {(const float*)d_in[10], (const float*)d_in[14], (const float*)d_in[18]};
    const float* emg_b[3]  = {(const float*)d_in[11], (const float*)d_in[15], (const float*)d_in[19]};
    const float* emg_gs[3] = {(const float*)d_in[12], (const float*)d_in[16], (const float*)d_in[20]};
    const float* emg_gb[3] = {(const float*)d_in[13], (const float*)d_in[17], (const float*)d_in[21]};
    const float* imu_w[3]  = {(const float*)d_in[22], (const float*)d_in[26], (const float*)d_in[30]};
    const float* imu_b[3]  = {(const float*)d_in[23], (const float*)d_in[27], (const float*)d_in[31]};
    const float* imu_gs[3] = {(const float*)d_in[24], (const float*)d_in[28], (const float*)d_in[32]};
    const float* imu_gb[3] = {(const float*)d_in[25], (const float*)d_in[29], (const float*)d_in[33]};
    const float* lstm_Wih[3] = {(const float*)d_in[34], (const float*)d_in[38], (const float*)d_in[42]};
    const float* lstm_Whh[3] = {(const float*)d_in[35], (const float*)d_in[39], (const float*)d_in[43]};
    const float* lstm_bih[3] = {(const float*)d_in[36], (const float*)d_in[40], (const float*)d_in[44]};
    const float* lstm_bhh[3] = {(const float*)d_in[37], (const float*)d_in[41], (const float*)d_in[45]};
    const float* hw1 = (const float*)d_in[46];
    const float* hb1 = (const float*)d_in[47];
    const float* hw2 = (const float*)d_in[48];
    const float* hb2 = (const float*)d_in[49];
    (void)in_sizes; (void)n_in; (void)out_size; (void)ws_size;

    char* p = (char*)d_ws;
    auto carve = [&](size_t bytes) -> void* {
        void* r = (void*)p;
        p += (bytes + 255) & ~(size_t)255;
        return r;
    };
    float*    ctx_y   = (float*)   carve((size_t)BB * CTXC * TT * 4);
    float*    rbar    = (float*)   carve((size_t)BB * CTXC * 4);
    float*    w_hard  = (float*)   carve((size_t)BB * EE * 4);
    _Float16* xe16    = (_Float16*)carve((size_t)BB * 16 * TT * 2);
    _Float16* xi16    = (_Float16*)carve((size_t)BB * 72 * TT * 2);
    float*    convbuf = (float*)   carve((size_t)EE * BB * 128 * TT * 4);
    _Float16* act1    = (_Float16*)carve((size_t)EE * BB * 32 * TT * 2);
    _Float16* act2    = (_Float16*)carve((size_t)EE * BB * 64 * TT * 2);
    _Float16* stacked = (_Float16*)carve((size_t)EE * BB * 256 * TT * 2);
    _Float16* hbufA   = (_Float16*)carve((size_t)BB * TT * 256 * 2);
    _Float16* hbufB   = (_Float16*)carve((size_t)BB * TT * 256 * 2);
    _Float16* Wih16   = (_Float16*)carve((size_t)2 * 512 * 256 * 2);
    _Float16* Whh16   = (_Float16*)carve((size_t)2 * 512 * 128 * 2);
    float*    xproj   = (float*)   carve((size_t)2 * 8192 * 512 * 4);
    // pre-cast, row-padded f16 conv weights
    _Float16* we16[3], *wi16[3];
    we16[0] = (_Float16*)carve((size_t)EE * 32  * 96  * 2);   // Klen 80  -> 96
    we16[1] = (_Float16*)carve((size_t)EE * 64  * 160 * 2);   // Klen 160 -> 160
    we16[2] = (_Float16*)carve((size_t)EE * 128 * 320 * 2);   // Klen 320 -> 320
    wi16[0] = (_Float16*)carve((size_t)EE * 32  * 384 * 2);   // Klen 360 -> 384
    wi16[1] = (_Float16*)carve((size_t)EE * 64  * 160 * 2);
    wi16[2] = (_Float16*)carve((size_t)EE * 128 * 320 * 2);

    // ---- context + gate
    k_ctx_conv<<<(BB * CTXC * TT + 255) / 256, 256, 0, stream>>>(x_emg, x_imu, ctx_cw, ctx_cb, ctx_y);
    k_ctx_gn<<<BB * 8, 256, 0, stream>>>(ctx_y, ctx_gs, ctx_gb, rbar);
    k_gate<<<1, 32, 0, stream>>>(rbar, ctx_lw, ctx_lb, gate_w, gate_b, w_hard);

    // ---- casts (inputs + all conv weights)
    k_cast_f16<<<512, 256, 0, stream>>>(x_emg, xe16, (long long)BB * 16 * TT);
    k_cast_f16<<<2304, 256, 0, stream>>>(x_imu, xi16, (long long)BB * 72 * TT);
    k_cast_pad<<<512, 256, 0, stream>>>(emg_w[0], we16[0], EE * 32, 80, 96);
    k_cast_pad<<<1024, 256, 0, stream>>>(emg_w[1], we16[1], EE * 64, 160, 160);
    k_cast_pad<<<2048, 256, 0, stream>>>(emg_w[2], we16[2], EE * 128, 320, 320);
    k_cast_pad<<<1024, 256, 0, stream>>>(imu_w[0], wi16[0], EE * 32, 360, 384);
    k_cast_pad<<<1024, 256, 0, stream>>>(imu_w[1], wi16[1], EE * 64, 160, 160);
    k_cast_pad<<<2048, 256, 0, stream>>>(imu_w[2], wi16[2], EE * 128, 320, 320);

    // ---- EMG experts: 16->32->64->128
    k_conv_wmma<16, 32><<<EE * BB * 2 * 32 / 8, 256, 0, stream>>>(xe16, 0LL, we16[0], emg_b[0], convbuf);
    k_gn_relu<<<EE * BB * GG, 256, 0, stream>>>(convbuf, emg_gs[0], emg_gb[0], act1, 32, 32, 0);
    k_conv_wmma<32, 64><<<EE * BB * 4 * 32 / 8, 256, 0, stream>>>(act1, (long long)BB * 32 * TT, we16[1], emg_b[1], convbuf);
    k_gn_relu<<<EE * BB * GG, 256, 0, stream>>>(convbuf, emg_gs[1], emg_gb[1], act2, 64, 64, 0);
    k_conv_wmma<64, 128><<<EE * BB * 8 * 32 / 8, 256, 0, stream>>>(act2, (long long)BB * 64 * TT, we16[2], emg_b[2], convbuf);
    k_gn_relu<<<EE * BB * GG, 256, 0, stream>>>(convbuf, emg_gs[2], emg_gb[2], stacked, 128, 256, 0);

    // ---- IMU experts: 72->32->64->128 (reuse act buffers)
    k_conv_wmma<72, 32><<<EE * BB * 2 * 32 / 8, 256, 0, stream>>>(xi16, 0LL, wi16[0], imu_b[0], convbuf);
    k_gn_relu<<<EE * BB * GG, 256, 0, stream>>>(convbuf, imu_gs[0], imu_gb[0], act1, 32, 32, 0);
    k_conv_wmma<32, 64><<<EE * BB * 4 * 32 / 8, 256, 0, stream>>>(act1, (long long)BB * 32 * TT, wi16[1], imu_b[1], convbuf);
    k_gn_relu<<<EE * BB * GG, 256, 0, stream>>>(convbuf, imu_gs[1], imu_gb[1], act2, 64, 64, 0);
    k_conv_wmma<64, 128><<<EE * BB * 8 * 32 / 8, 256, 0, stream>>>(act2, (long long)BB * 64 * TT, wi16[2], imu_b[2], convbuf);
    k_gn_relu<<<EE * BB * GG, 256, 0, stream>>>(convbuf, imu_gs[2], imu_gb[2], stacked, 128, 256, 128);

    // ---- gated combine -> hbufA [B][T][256] f16
    k_combine<<<(BB * 256 * (TT / 8)) / 256, 256, 0, stream>>>(stacked, w_hard, hbufA);

    // ---- 3 bidirectional LSTM layers
    _Float16* hin = hbufA;
    _Float16* hout = hbufB;
    for (int layer = 0; layer < 3; layer++) {
        k_cast_f16<<<1024, 256, 0, stream>>>(lstm_Wih[layer], Wih16, (long long)2 * 512 * 256);
        k_cast_f16<<<512, 256, 0, stream>>>(lstm_Whh[layer], Whh16, (long long)2 * 512 * 128);
        k_xproj<<<4096, 256, 0, stream>>>(hin, Wih16, lstm_bih[layer], lstm_bhh[layer], xproj);
        k_lstm_rec<<<2, 512, 0, stream>>>(xproj, Whh16, hout);
        _Float16* t = hin; hin = hout; hout = t;
    }

    // ---- head
    k_head<<<1, 256, 0, stream>>>(hin, hw1, hb1, hw2, hb2, (float*)d_out);
}
